// SparseSpikingAttention_84155589198554
// MI455X (gfx1250) — compile-verified, matrix operand order checked by reading, not run on hardware
//
#include <hip/hip_runtime.h>
#include <hip/hip_bf16.h>
#include <cstdint>

typedef __attribute__((ext_vector_type(16))) _Float16 v16h;
typedef __attribute__((ext_vector_type(8)))  _Float16 v8h;
typedef __attribute__((ext_vector_type(8)))  float    v8f;
typedef __attribute__((ext_vector_type(4)))  unsigned v4u;
typedef __attribute__((ext_vector_type(8)))  unsigned v8u;

#define WMMA_F16(a,b,c) __builtin_amdgcn_wmma_f32_16x16x32_f16(false,(a),false,(b),(short)0,(c),false,false)

static constexpr int Bsz = 2, T = 8, S = 512, D = 1024, H = 16, HD = 64;
static constexpr int M_ROWS = Bsz * T * S;  // 8192

// ---------------------------------------------------------------------------
// Fragment load: 16 rows x 32 k of f16 from a row-major matrix (ld elements).
// CDNA5 16-bit A/B layout: lanes 0-15 hold K {0..7,16..23} of row=lane,
// lanes 16-31 hold K {8..15,24..31} of row=lane-16. Two b128 loads per lane.
// ---------------------------------------------------------------------------
__device__ inline v16h load_frag(const _Float16* __restrict__ base, int ld) {
  const int lane = threadIdx.x & 31;
  const int row  = lane & 15;
  const int koff = (lane < 16) ? 0 : 8;
  const _Float16* p = base + row * ld + koff;
  v8h lo = *(const v8h*)(p);
  v8h hi = *(const v8h*)(p + 16);
  v16h r;
#pragma unroll
  for (int i = 0; i < 8; ++i) { r[i] = lo[i]; r[i + 8] = hi[i]; }
  return r;
}

// Async global->LDS copy (ASYNCcnt).
__device__ inline void async_b128_to_lds(uint32_t lds_addr, const void* g) {
  asm volatile("global_load_async_to_lds_b128 %0, %1, off"
               :: "v"(lds_addr), "v"((uint64_t)(uintptr_t)g) : "memory");
}
__device__ inline void wait_asynccnt0() {
  asm volatile("s_wait_asynccnt 0x0" ::: "memory");
}

// Tensor Data Mover: 1-D tile of nelem f16 elements, global -> LDS (TENSORcnt).
// D# built per CDNA5 ISA (count=1, data_size=2B,
// type=2 "image"; tile_dim0=tensor_dim0=nelem, tensor_dim1=1).
__device__ inline void tdm_load_f16_1d(uint32_t lds_addr, const void* gptr, uint32_t nelem) {
  const uint64_t ga = (uint64_t)(uintptr_t)gptr;
  v4u g0;
  g0[0] = 1u;                                                   // count=1 (valid user D#)
  g0[1] = lds_addr;                                             // lds_addr [63:32]
  g0[2] = (uint32_t)(ga & 0xFFFFFFFFu);                         // global_addr [95:64]
  g0[3] = ((uint32_t)(ga >> 32) & 0x01FFFFFFu) | 0x80000000u;   // addr[56:32] | type=2
  v8u g1;
  g1[0] = 0x00010000u;                                          // data_size=1 (2 bytes)
  g1[1] = (nelem & 0xFFFFu) << 16;                              // tensor_dim0[15:0]
  g1[2] = (nelem >> 16) | (1u << 16);                           // dim0[31:16] | tensor_dim1=1
  g1[3] = (nelem & 0xFFFFu) << 16;                              // tile_dim0
  g1[4] = 0u;                                                   // tile_dim1/2 unused
  g1[5] = nelem;                                                // tensor_dim0_stride[31:0]
  g1[6] = 0u;
  g1[7] = 0u;
  asm volatile("tensor_load_to_lds %0, %1" :: "s"(g0), "s"(g1) : "memory");
}

// ---------------------------------------------------------------------------
// 1) f32 -> f16 conversion
// ---------------------------------------------------------------------------
__global__ void k_f32_to_f16(const float* __restrict__ src, _Float16* __restrict__ dst, int n) {
  int i = blockIdx.x * blockDim.x + threadIdx.x;
  if (i < n) dst[i] = (_Float16)src[i];
}

// ---------------------------------------------------------------------------
// 2) QKV projection: out = x @ W^T + b. One wave per 16x64 tile, explicitly
//    double-buffered: next k-step's 5 fragments load while the current 4
//    WMMAs execute, so waits are partial (loadcnt<=10) instead of full drains.
// ---------------------------------------------------------------------------
__global__ void k_qkv_gemm(const _Float16* __restrict__ xh,
                           const _Float16* __restrict__ Wq,
                           const _Float16* __restrict__ Wk,
                           const _Float16* __restrict__ Wv,
                           const float* __restrict__ bq,
                           const float* __restrict__ bk,
                           const float* __restrict__ bv,
                           _Float16* __restrict__ Qh,
                           _Float16* __restrict__ Kh,
                           _Float16* __restrict__ Vt) {
  const int n0 = blockIdx.x * 64;
  const int m0 = blockIdx.y * 16;
  const int which = blockIdx.z;
  const _Float16* W = (which == 0) ? Wq : (which == 1) ? Wk : Wv;
  const float* bias  = (which == 0) ? bq : (which == 1) ? bk : bv;
  const int lane = threadIdx.x & 31;

  const _Float16* Ab = xh + (size_t)m0 * D;
  const _Float16* Bb = W + (size_t)n0 * D;

  v8f acc0 = {}, acc1 = {}, acc2 = {}, acc3 = {};
  // prologue: fragments for k=0
  v16h a  = load_frag(Ab, D);
  v16h b0 = load_frag(Bb + (size_t)0 * 16 * D, D);
  v16h b1 = load_frag(Bb + (size_t)1 * 16 * D, D);
  v16h b2 = load_frag(Bb + (size_t)2 * 16 * D, D);
  v16h b3 = load_frag(Bb + (size_t)3 * 16 * D, D);
#pragma unroll 2
  for (int k = 0; k < D; k += 32) {
    const int kn = (k + 32) & (D - 1);  // wraps to 0 on last iter (discarded)
    v16h an  = load_frag(Ab + kn, D);
    v16h bn0 = load_frag(Bb + (size_t)0 * 16 * D + kn, D);
    v16h bn1 = load_frag(Bb + (size_t)1 * 16 * D + kn, D);
    v16h bn2 = load_frag(Bb + (size_t)2 * 16 * D + kn, D);
    v16h bn3 = load_frag(Bb + (size_t)3 * 16 * D + kn, D);
    __builtin_prefetch(Ab + kn + 128, 0, 1);
    acc0 = WMMA_F16(a, b0, acc0);
    acc1 = WMMA_F16(a, b1, acc1);
    acc2 = WMMA_F16(a, b2, acc2);
    acc3 = WMMA_F16(a, b3, acc3);
    a = an; b0 = bn0; b1 = bn1; b2 = bn2; b3 = bn3;
  }

  const int h  = n0 >> 6;            // 64-wide tile == one head
  const int mb = m0 + ((lane < 16) ? 0 : 8);
  const v8f accs[4] = {acc0, acc1, acc2, acc3};
#pragma unroll
  for (int j = 0; j < 4; ++j) {
    const int hd = 16 * j + (lane & 15);
    const float bval = bias[n0 + 16 * j + (lane & 15)];
#pragma unroll
    for (int r = 0; r < 8; ++r) {
      const int m = mb + r;
      const int bt = m >> 9, s = m & 511;
      _Float16 v = (_Float16)(accs[j][r] + bval);
      if (which == 0)      Qh[(((size_t)bt * H + h) * S + s) * HD + hd] = v;
      else if (which == 1) Kh[(((size_t)bt * H + h) * S + s) * HD + hd] = v;
      else                 Vt[(((size_t)bt * H + h) * HD + hd) * S + s] = v;
    }
  }
}

// ---------------------------------------------------------------------------
// 3) Attention per (b,t,h, 16-row q tile). 4 waves (128 threads).
//    LDS: scores f32 16x512 | attn f16 16x512 | reductions | Qtile | V 64x512
//    V staged by the Tensor Data Mover (TENSORcnt), Q tile by async-to-LDS
//    (ASYNCcnt), waited on first; only V overlaps the Q*K^T WMMAs.
// ---------------------------------------------------------------------------
#define SC_OFF   0
#define AT_OFF   (16 * 512 * 4)            // 32768
#define RED_OFF  (AT_OFF + 16 * 512 * 2)   // 49152
#define RM_OFF   (RED_OFF + 128 * 4)       // 49664
#define RS_OFF   (RM_OFF + 16 * 4)         // 49728
#define QT_OFF   (RS_OFF + 16 * 4)         // 49792 (128B aligned)
#define V_OFF    (QT_OFF + 16 * 64 * 2)    // 51840 (128B aligned)
#define SMEM_BYTES (V_OFF + HD * S * 2)    // 117376

__global__ void k_attention(const _Float16* __restrict__ Qh,
                            const _Float16* __restrict__ Kh,
                            const _Float16* __restrict__ Vt,
                            const float* __restrict__ tw,
                            float* __restrict__ att) {
  extern __shared__ char smem[];
  float*    sc   = (float*)(smem + SC_OFF);
  _Float16* at   = (_Float16*)(smem + AT_OFF);
  float*    red  = (float*)(smem + RED_OFF);
  float*    rmax = (float*)(smem + RM_OFF);
  float*    rsum = (float*)(smem + RS_OFF);
  _Float16* qt   = (_Float16*)(smem + QT_OFF);
  _Float16* vlds = (_Float16*)(smem + V_OFF);

  const int tid  = threadIdx.x;
  const int lane = tid & 31;
  const int wave = tid >> 5;
  const int q0   = blockIdx.x * 16;
  const int bth  = blockIdx.y;
  const int h    = bth & (H - 1);
  const int bt   = bth >> 4;
  const int t    = bt & (T - 1);

  const _Float16* Q = Qh + (size_t)bth * S * HD;
  const _Float16* K = Kh + (size_t)bth * S * HD;
  const _Float16* V = Vt + (size_t)bth * HD * S;

  // TDM: stage V (64KB, contiguous) into LDS; issued once, by wave 0.
  if (wave == 0) {
    tdm_load_f16_1d((uint32_t)(uintptr_t)vlds, V, HD * S);
  }
  // Async: stage the Q tile (16x64 f16 = 2KB contiguous), one b128 per thread.
  async_b128_to_lds((uint32_t)(uintptr_t)qt + (uint32_t)tid * 16u,
                    (const char*)(Q + q0 * HD) + tid * 16);
  wait_asynccnt0();
  __syncthreads();

  // scores = (Q K^T) * tw/sqrt(HD); wave w covers k-tiles [w*8, w*8+8),
  // software-pipelined: next tile's K fragments load during current WMMAs.
  const float fac = tw[t * H + h] * 0.125f;  // 1/sqrt(64)
  v16h aq0 = load_frag(qt + 0, HD);
  v16h aq1 = load_frag(qt + 32, HD);
  v16h kb0 = load_frag(K + (wave * 8) * 16 * HD + 0, HD);
  v16h kb1 = load_frag(K + (wave * 8) * 16 * HD + 32, HD);
#pragma unroll
  for (int j = 0; j < 8; ++j) {
    const int kt = (wave * 8 + j) * 16;
    const int ktn = (wave * 8 + ((j + 1) & 7)) * 16;  // wraps (discarded)
    v16h kn0 = load_frag(K + ktn * HD + 0, HD);
    v16h kn1 = load_frag(K + ktn * HD + 32, HD);
    v8f acc = {};
    acc = WMMA_F16(aq0, kb0, acc);
    acc = WMMA_F16(aq1, kb1, acc);
    kb0 = kn0; kb1 = kn1;
    const int col = kt + (lane & 15);
    const int mb  = (lane < 16) ? 0 : 8;
#pragma unroll
    for (int r = 0; r < 8; ++r) sc[(mb + r) * S + col] = acc[r] * fac;
  }
  __syncthreads();

  // Row max (16 rows x 8 segments of 64)
  const int row = tid >> 3, seg = tid & 7;
  {
    float m = -3.4e38f;
    const float* p = sc + row * S + seg * 64;
#pragma unroll 8
    for (int i = 0; i < 64; ++i) m = fmaxf(m, p[i]);
    red[tid] = m;
  }
  __syncthreads();
  if (tid < 16) {
    float m = red[tid * 8];
#pragma unroll
    for (int i = 1; i < 8; ++i) m = fmaxf(m, red[tid * 8 + i]);
    rmax[tid] = m;
  }
  __syncthreads();

  // Spike (temperature-scaled sigmoid, hard threshold) + row sum
  const float temp = 1.0f + 0.5f * cosf((float)t * (3.14159265358979f / 8.0f));
  {
    const float mx = rmax[row];
    float ssum = 0.f;
    const float* p = sc + row * S + seg * 64;
    _Float16* q = at + row * S + seg * 64;
#pragma unroll 4
    for (int i = 0; i < 64; ++i) {
      const float z    = 5.0f * (p[i] - mx) / temp;
      const float prob = 1.0f / (1.0f + expf(-z));
      const float spk  = (prob > 0.5f) ? 1.0f : 0.0f;
      q[i] = (_Float16)spk;
      ssum += spk;
    }
    red[tid] = ssum;
  }
  __syncthreads();
  if (tid < 16) {
    float s = 0.f;
#pragma unroll
    for (int i = 0; i < 8; ++i) s += red[tid * 8 + i];
    rsum[tid] = s;
  }
  __syncthreads();
  {
    const float inv = 1.0f / (rsum[row] + 1e-8f);
    _Float16* q = at + row * S + seg * 64;
#pragma unroll 4
    for (int i = 0; i < 64; ++i) q[i] = (_Float16)((float)q[i] * inv);
  }

  // TDM data must be resident before the second GEMM: issuing wave drains
  // its TENSORcnt, then the workgroup barrier publishes LDS to all waves.
  if (wave == 0) __builtin_amdgcn_s_wait_tensorcnt(0);
  __syncthreads();

  // attended = attn @ V : wave w -> hd tile w (16 cols), contract over S=512
  {
    const int n0 = wave * 16;
    v8f acc = {};
#pragma unroll 4
    for (int kk = 0; kk < S; kk += 32) {
      v16h a = load_frag(at + kk, S);
      v16h b = load_frag(vlds + n0 * S + kk, S);
      acc = WMMA_F16(a, b, acc);
    }
    const int n  = n0 + (lane & 15);
    const int mb = q0 + ((lane < 16) ? 0 : 8);
#pragma unroll
    for (int r = 0; r < 8; ++r)
      att[((size_t)bth * S + (mb + r)) * HD + n] = acc[r];
  }
}

// ---------------------------------------------------------------------------
// 4) means[t,h,hd] = mean over (b,s) of attended
// ---------------------------------------------------------------------------
__global__ void k_means(const float* __restrict__ att, float* __restrict__ means) {
  __shared__ float part[256];
  const int th = blockIdx.x;  // t*H + h
  const int t = th / H, h = th % H;
  const int tid = threadIdx.x;
  const int hd = tid & 63, chunk = tid >> 6;
  float s = 0.f;
  for (int r = chunk * 256; r < chunk * 256 + 256; ++r) {
    const int b = r >> 9, sidx = r & 511;
    s += att[((((size_t)(b * T + t) * H + h) * S + sidx)) * HD + hd];
  }
  part[tid] = s;
  __syncthreads();
  if (tid < 64) {
    const float tot = part[tid] + part[tid + 64] + part[tid + 128] + part[tid + 192];
    means[(size_t)th * HD + hd] = tot * (1.0f / 1024.0f);
  }
}

// ---------------------------------------------------------------------------
// 5) EMA scan over t (1024 independent (h,hd) chains)
// ---------------------------------------------------------------------------
__global__ void k_ema(const float* __restrict__ means, const float* __restrict__ decay,
                      const float* __restrict__ mem0, float* __restrict__ mems) {
  const int i = blockIdx.x * blockDim.x + threadIdx.x;
  if (i >= H * HD) return;
  const float d = decay[0];
  float m = mem0[i];
  for (int t = 0; t < T; ++t) {
    m = d * m + (1.0f - d) * means[t * H * HD + i];
    mems[t * H * HD + i] = m;
  }
}

// ---------------------------------------------------------------------------
// 6) Build f16 operand for output projection: (b,t,s,(h,hd)) + 0.1*mems
// ---------------------------------------------------------------------------
__global__ void k_build_o(const float* __restrict__ att, const float* __restrict__ mems,
                          _Float16* __restrict__ Oh) {
  const int i = blockIdx.x * blockDim.x + threadIdx.x;
  if (i >= M_ROWS * D) return;
  const int m = i >> 10, c = i & 1023;
  const int bt = m >> 9, s = m & 511;
  const int h = c >> 6, hd = c & 63;
  const int t = bt & (T - 1);
  const float v = att[(((size_t)bt * H + h) * S + s) * HD + hd]
                + 0.1f * mems[(t * H + h) * HD + hd];
  Oh[i] = (_Float16)v;
}

// ---------------------------------------------------------------------------
// 7) Output projection: out = O @ Wo^T + bo (f32), 16x64 tile per wave,
//    double-buffered like the QKV GEMM.
// ---------------------------------------------------------------------------
__global__ void k_out_gemm(const _Float16* __restrict__ Oh, const _Float16* __restrict__ Wo,
                           const float* __restrict__ bo, float* __restrict__ out) {
  const int n0 = blockIdx.x * 64, m0 = blockIdx.y * 16;
  const int lane = threadIdx.x & 31;
  const _Float16* Ab = Oh + (size_t)m0 * D;
  const _Float16* Bb = Wo + (size_t)n0 * D;

  v8f acc0 = {}, acc1 = {}, acc2 = {}, acc3 = {};
  v16h a  = load_frag(Ab, D);
  v16h b0 = load_frag(Bb + (size_t)0 * 16 * D, D);
  v16h b1 = load_frag(Bb + (size_t)1 * 16 * D, D);
  v16h b2 = load_frag(Bb + (size_t)2 * 16 * D, D);
  v16h b3 = load_frag(Bb + (size_t)3 * 16 * D, D);
#pragma unroll 2
  for (int k = 0; k < D; k += 32) {
    const int kn = (k + 32) & (D - 1);
    v16h an  = load_frag(Ab + kn, D);
    v16h bn0 = load_frag(Bb + (size_t)0 * 16 * D + kn, D);
    v16h bn1 = load_frag(Bb + (size_t)1 * 16 * D + kn, D);
    v16h bn2 = load_frag(Bb + (size_t)2 * 16 * D + kn, D);
    v16h bn3 = load_frag(Bb + (size_t)3 * 16 * D + kn, D);
    __builtin_prefetch(Ab + kn + 128, 0, 1);
    acc0 = WMMA_F16(a, b0, acc0);
    acc1 = WMMA_F16(a, b1, acc1);
    acc2 = WMMA_F16(a, b2, acc2);
    acc3 = WMMA_F16(a, b3, acc3);
    a = an; b0 = bn0; b1 = bn1; b2 = bn2; b3 = bn3;
  }
  const int mb = m0 + ((lane < 16) ? 0 : 8);
  const v8f accs[4] = {acc0, acc1, acc2, acc3};
#pragma unroll
  for (int j = 0; j < 4; ++j) {
    const int n = n0 + 16 * j + (lane & 15);
    const float bv = bo[n];
#pragma unroll
    for (int r = 0; r < 8; ++r) out[(size_t)(mb + r) * D + n] = accs[j][r] + bv;
  }
}

// ---------------------------------------------------------------------------
extern "C" void kernel_launch(void* const* d_in, const int* in_sizes, int n_in,
                              void* d_out, int out_size, void* d_ws, size_t ws_size,
                              hipStream_t stream) {
  const float* x     = (const float*)d_in[0];
  const float* Wq    = (const float*)d_in[1];
  const float* bq    = (const float*)d_in[2];
  const float* Wk    = (const float*)d_in[3];
  const float* bk    = (const float*)d_in[4];
  const float* Wv    = (const float*)d_in[5];
  const float* bv    = (const float*)d_in[6];
  const float* Wo    = (const float*)d_in[7];
  const float* bo    = (const float*)d_in[8];
  const float* tw    = (const float*)d_in[9];
  const float* decay = (const float*)d_in[10];
  const float* mem0  = (const float*)d_in[11];
  float* out = (float*)d_out;

  char* ws = (char*)d_ws;
  size_t off = 0;
  auto alloc = [&](size_t bytes) {
    char* p = ws + off;
    off = (off + bytes + 255) & ~(size_t)255;
    return p;
  };
  _Float16* xh   = (_Float16*)alloc((size_t)M_ROWS * D * 2);  // 16 MB
  _Float16* Wqh  = (_Float16*)alloc((size_t)D * D * 2);
  _Float16* Wkh  = (_Float16*)alloc((size_t)D * D * 2);
  _Float16* Wvh  = (_Float16*)alloc((size_t)D * D * 2);
  _Float16* Woh  = (_Float16*)alloc((size_t)D * D * 2);
  _Float16* Qh   = (_Float16*)alloc((size_t)M_ROWS * D * 2);
  _Float16* Kh   = (_Float16*)alloc((size_t)M_ROWS * D * 2);
  _Float16* Vt   = (_Float16*)alloc((size_t)M_ROWS * D * 2);
  float*    att  = (float*)   alloc((size_t)M_ROWS * D * 4);  // 32 MB
  _Float16* Oh   = (_Float16*)alloc((size_t)M_ROWS * D * 2);
  float*    mean = (float*)   alloc((size_t)T * H * HD * 4);
  float*    mems = (float*)   alloc((size_t)T * H * HD * 4);

  const int nX = M_ROWS * D, nW = D * D;

  // 1) conversions
  k_f32_to_f16<<<(nX + 255) / 256, 256, 0, stream>>>(x,  xh,  nX);
  k_f32_to_f16<<<(nW + 255) / 256, 256, 0, stream>>>(Wq, Wqh, nW);
  k_f32_to_f16<<<(nW + 255) / 256, 256, 0, stream>>>(Wk, Wkh, nW);
  k_f32_to_f16<<<(nW + 255) / 256, 256, 0, stream>>>(Wv, Wvh, nW);
  k_f32_to_f16<<<(nW + 255) / 256, 256, 0, stream>>>(Wo, Woh, nW);

  // 2) QKV projection (z: 0=Q,1=K,2=V), 16x64 tile per wave
  k_qkv_gemm<<<dim3(D / 64, M_ROWS / 16, 3), 32, 0, stream>>>(
      xh, Wqh, Wkh, Wvh, bq, bk, bv, Qh, Kh, Vt);

  // 3) attention (4 waves, ~115KB dynamic LDS, TDM + async staging)
  k_attention<<<dim3(S / 16, Bsz * T * H), 128, SMEM_BYTES, stream>>>(Qh, Kh, Vt, tw, att);

  // 4) means  5) EMA
  k_means<<<T * H, 256, 0, stream>>>(att, mean);
  k_ema<<<4, 256, 0, stream>>>(mean, decay, mem0, mems);

  // 6) build O (+0.1*mems, f16)  7) output projection
  k_build_o<<<(nX + 255) / 256, 256, 0, stream>>>(att, mems, Oh);
  k_out_gemm<<<dim3(D / 64, M_ROWS / 16), 32, 0, stream>>>(Oh, Woh, bo, out);
}
